// BahdanauAttention_25228637896977
// MI455X (gfx1250) — matrix operand order repaired and checked by reading, not compiled, not run on hardware
//
#include <hip/hip_runtime.h>
#include <hip/hip_bf16.h>

#define Bc 64
#define Sc 2048
#define Hc 512
#define NEGV -1000000000.0f

typedef __attribute__((ext_vector_type(16))) _Float16 v16h;
typedef __attribute__((ext_vector_type(8)))  _Float16 h8;
typedef __attribute__((ext_vector_type(4)))  _Float16 h4;
typedef __attribute__((ext_vector_type(8)))  float    v8f;

// Fast tanh: use native transcendental tanh if the builtin exists, else
// tanh(x) = 1 - 2/(exp(2x)+1)  (v_exp_f32 + v_rcp_f32; correct limits at +-inf).
__device__ __forceinline__ float fast_tanh(float x) {
#if __has_builtin(__builtin_amdgcn_tanhf)
  return __builtin_amdgcn_tanhf(x);
#else
  float e = __expf(2.0f * x);
  return 1.0f - 2.0f * __builtin_amdgcn_rcpf(e + 1.0f);
#endif
}

// ---------------------------------------------------------------------------
// Kernel 1: W = f16(Wh + Ws), row-major (o, h)  -> K-contiguous B matrix
// ---------------------------------------------------------------------------
__global__ __launch_bounds__(256) void prep_w_kernel(
    const float* __restrict__ Wh, const float* __restrict__ Ws,
    _Float16* __restrict__ Wo) {
  int i = blockIdx.x * 256 + threadIdx.x;
  Wo[i] = (_Float16)(Wh[i] + Ws[i]);
}

// ---------------------------------------------------------------------------
// Kernel 2: fused GEMM (X @ W^T) + tanh + dot(v) + mask  -> score[B*S]
// One workgroup (8 waves) per 64-row tile. Each wave owns 64 N columns and
// all 4 m-blocks, so every B fragment feeds 4 WMMAs (4x B-load amortization).
// ---------------------------------------------------------------------------
__global__ __launch_bounds__(256) void score_kernel(
    const float* __restrict__ X, const int* __restrict__ mask,
    const _Float16* __restrict__ W, const float* __restrict__ v,
    float* __restrict__ score) {
  constexpr int RS = Hc + 8;                 // padded LDS row stride (halves)
  __shared__ _Float16 xs[64 * RS];           // 64 x 512 f16 x-tile (~65 KB)
  __shared__ float    vsh[Hc];
  __shared__ float    ssc[64];

  const int tid = threadIdx.x;
  const size_t R0 = (size_t)blockIdx.x * 64; // first global row of tile

  // Stage x tile (f32 -> f16) into LDS: float4 loads, packed b64 LDS stores.
  for (int g = tid; g < (64 * Hc) / 4; g += 256) {
    int r = g >> 7;                          // 128 float4 groups per row
    int c = (g & 127) * 4;
    float4 xv = *(const float4*)&X[(R0 + r) * Hc + c];
    h4 o;
    o[0] = (_Float16)xv.x; o[1] = (_Float16)xv.y;
    o[2] = (_Float16)xv.z; o[3] = (_Float16)xv.w;
    *(h4*)&xs[r * RS + c] = o;               // ds_store_b64
  }
  vsh[tid] = v[tid];
  vsh[tid + 256] = v[tid + 256];
  if (tid < 64) ssc[tid] = 0.0f;
  __syncthreads();

  const int lane  = tid & 31;
  const int wave  = tid >> 5;
  const int half  = lane >> 4;     // 0: lanes 0-15, 1: lanes 16-31
  const int ln    = lane & 15;
  const int nbase = wave * 64;     // this wave's first output column

  v8f acc[4][4] = {};              // [m-block][n-tile] accumulators (128 VGPRs)
  const int aoff = (half ? 8 : 0);

  for (int k0 = 0; k0 < Hc; k0 += 32) {
    // A fragments for the 4 m-blocks (16x32 f16 each):
    // lanes<16 hold K {0..7,16..23}, lanes>=16 hold K {8..15,24..31}
    v16h a[4];
#pragma unroll
    for (int m = 0; m < 4; ++m) {
      const _Float16* ap = &xs[(m * 16 + ln) * RS + k0 + aoff];
      ((h8*)&a[m])[0] = *(const h8*)(ap);        // ds_load_b128
      ((h8*)&a[m])[1] = *(const h8*)(ap + 16);   // ds_load_b128
    }
#pragma unroll
    for (int nt = 0; nt < 4; ++nt) {
      // B fragment (32x16 f16): lane = column n, lanes<16 K {0..7,16..23}, >=16 K {8..15,24..31}
      const _Float16* bp = W + (size_t)(nbase + nt * 16 + ln) * Hc + k0 + half * 8;
      v16h bfr;
      ((h8*)&bfr)[0] = *(const h8*)(bp);         // global_load_b128 (L2-hot)
      ((h8*)&bfr)[1] = *(const h8*)(bp + 16);
#pragma unroll
      for (int m = 0; m < 4; ++m)
        acc[m][nt] = __builtin_amdgcn_wmma_f32_16x16x32_f16(
            false, a[m], false, bfr, (short)0, acc[m][nt], false, false);
    }
  }

  // Epilogue: score[m-row] += sum_n tanh(D[m][n]) * v[n]
  // D layout: lane holds column N = ln, VGPR r holds row (r + 8*half).
#pragma unroll
  for (int m = 0; m < 4; ++m) {
    float pr[8] = {0, 0, 0, 0, 0, 0, 0, 0};
#pragma unroll
    for (int nt = 0; nt < 4; ++nt) {
      float vn = vsh[nbase + nt * 16 + ln];
#pragma unroll
      for (int r = 0; r < 8; ++r)
        pr[r] += fast_tanh(acc[m][nt][r]) * vn;
    }
#pragma unroll
    for (int r = 0; r < 8; ++r) {
      float t = pr[r];
      t += __shfl_xor(t, 1, 32);
      t += __shfl_xor(t, 2, 32);
      t += __shfl_xor(t, 4, 32);
      t += __shfl_xor(t, 8, 32);             // reduce over the 16 N lanes
      if (ln == 0) atomicAdd(&ssc[m * 16 + half * 8 + r], t);  // ds_add_f32
    }
  }
  __syncthreads();

  if (tid < 64) {
    size_t row = R0 + tid;
    score[row] = (mask[row] == 0) ? NEGV : ssc[tid];
  }
}

// ---------------------------------------------------------------------------
// Kernel 3: per-batch softmax over S; also zero-init context output.
// ---------------------------------------------------------------------------
__global__ __launch_bounds__(256) void softmax_kernel(
    const float* __restrict__ score, float* __restrict__ ctx,
    float* __restrict__ attn) {
  __shared__ float red[256];
  const int b = blockIdx.x, tid = threadIdx.x;

  ctx[b * Hc + tid] = 0.0f;                  // zero context (atomics follow later)
  ctx[b * Hc + 256 + tid] = 0.0f;

  const float* s = score + (size_t)b * Sc;
  float lv[8];
  float m = -3.4e38f;
#pragma unroll
  for (int j = 0; j < 8; ++j) { lv[j] = s[tid + j * 256]; m = fmaxf(m, lv[j]); }
  red[tid] = m;
  __syncthreads();
  for (int off = 128; off > 0; off >>= 1) {
    if (tid < off) red[tid] = fmaxf(red[tid], red[tid + off]);
    __syncthreads();
  }
  m = red[0];
  __syncthreads();

  float sum = 0.0f;
#pragma unroll
  for (int j = 0; j < 8; ++j) { lv[j] = __expf(lv[j] - m); sum += lv[j]; }
  red[tid] = sum;
  __syncthreads();
  for (int off = 128; off > 0; off >>= 1) {
    if (tid < off) red[tid] += red[tid + off];
    __syncthreads();
  }
  const float inv = 1.0f / red[0];
#pragma unroll
  for (int j = 0; j < 8; ++j)
    attn[(size_t)b * Sc + tid + j * 256] = lv[j] * inv;
}

// ---------------------------------------------------------------------------
// Kernel 4: context[b,h] = sum_s attn[b,s] * X[b,s,h]   (memory-bound GEMV)
// grid = (B, S/256); each thread owns 2 h-columns, atomic accumulate.
// ---------------------------------------------------------------------------
__global__ __launch_bounds__(256) void context_kernel(
    const float* __restrict__ X, const float* __restrict__ attn,
    float* __restrict__ ctx) {
  __shared__ float wsm[256];
  const int b = blockIdx.x, chunk = blockIdx.y, tid = threadIdx.x;
  wsm[tid] = attn[(size_t)b * Sc + chunk * 256 + tid];
  __syncthreads();

  const float* xbase = X + ((size_t)(b * Sc + chunk * 256)) * Hc + tid * 2;
  float ax = 0.0f, ay = 0.0f;
#pragma unroll 4
  for (int i = 0; i < 256; ++i) {
    float2 xv = *(const float2*)(xbase + (size_t)i * Hc);
    float w = wsm[i];
    ax = fmaf(w, xv.x, ax);
    ay = fmaf(w, xv.y, ay);
  }
  atomicAdd(&ctx[b * Hc + tid * 2 + 0], ax);
  atomicAdd(&ctx[b * Hc + tid * 2 + 1], ay);
}

// ---------------------------------------------------------------------------
extern "C" void kernel_launch(void* const* d_in, const int* in_sizes, int n_in,
                              void* d_out, int out_size, void* d_ws, size_t ws_size,
                              hipStream_t stream) {
  const float* X    = (const float*)d_in[0];  // (B,S,H)
  const int*   mask = (const int*)d_in[1];    // (B,S)
  const float* Wh   = (const float*)d_in[2];  // (H,H)
  const float* Ws   = (const float*)d_in[3];  // (H,H)
  const float* v    = (const float*)d_in[4];  // (H,)

  float* ctx  = (float*)d_out;                // (B,H)
  float* attn = (float*)d_out + Bc * Hc;      // (B,S)

  _Float16* Wf    = (_Float16*)d_ws;                                   // 512 KB
  float*    score = (float*)((char*)d_ws + (size_t)Hc * Hc * sizeof(_Float16)); // 512 KB

  prep_w_kernel<<<(Hc * Hc) / 256, 256, 0, stream>>>(Wh, Ws, Wf);
  score_kernel<<<(Bc * Sc) / 64, 256, 0, stream>>>(X, mask, Wf, v, score);
  softmax_kernel<<<Bc, 256, 0, stream>>>(score, ctx, attn);
  context_kernel<<<dim3(Bc, Sc / 256), 256, 0, stream>>>(X, attn, ctx);
}
